// ODERecurrentPointProcess_16690242913098
// MI455X (gfx1250) — matrix operand order repaired and checked by reading, not compiled, not run on hardware
//
#include <hip/hip_runtime.h>
#include <hip/hip_bf16.h>

typedef __attribute__((ext_vector_type(16))) __bf16 v16bf;
typedef __attribute__((ext_vector_type(8)))  __bf16 v8bf;
typedef __attribute__((ext_vector_type(8)))  float  v8f;

#define DD   256
#define TT   512
#define BB   128
#define KT8  8        // K tiles for K=256 (K-step = 32)
#define ARS  264      // padded row stride (halfs) for LDS activation buffers: 528B == 132 dw
                      // -> row banks stride 4 mod 64: conflict-free A-fragment reads

// ---------------- WMMA helpers (base-pointer style: per-kt offsets are immediates) --------

__device__ __forceinline__ v8f wmma_bf(v16bf a, v16bf b, v8f c) {
  return __builtin_amdgcn_wmma_f32_16x16x32_bf16(false, a, false, b, (short)0, c, false, false);
}

__device__ __forceinline__ v16bf cat8(v8bf lo, v8bf hi) {
  return __builtin_shufflevector(lo, hi, 0,1,2,3,4,5,6,7,8,9,10,11,12,13,14,15);
}

// B operand: tile (nt,kt) = 512 halfs as two 256-half chunks (bank-conflict-free in LDS,
// coalesced in L2). Base is fixed per (matrix, nt, lane); kt -> constant byte offset.
__device__ __forceinline__ const __bf16* b_base(const __bf16* P, int nt, int lane) {
  return P + (((size_t)nt * KT8) << 9) + lane * 8;
}
__device__ __forceinline__ v16bf ld_b_at(const __bf16* base, int kt) {
  v8bf lo = *(const v8bf*)(base + kt * 512);
  v8bf hi = *(const v8bf*)(base + kt * 512 + 256);
  return cat8(lo, hi);
}

// A operand from row-major bf16 [16][rs] LDS buffer (ISA interleave):
// lane<16: row m=lane, K {k0..k0+7, k0+16..k0+23}; lane>=16: K {+8,+24}.
__device__ __forceinline__ const __bf16* a_lds_base(const __bf16* buf, int lane, int rs) {
  return buf + (lane & 15) * rs + (lane >> 4) * 8;
}
__device__ __forceinline__ v16bf ld_a_at(const __bf16* base, int kt) {
  v8bf lo = *(const v8bf*)(base + kt * 32);
  v8bf hi = *(const v8bf*)(base + kt * 32 + 16);
  return cat8(lo, hi);
}

// A operand from packed global tiles (16 contiguous halfs per lane).
__device__ __forceinline__ const __bf16* a_pk_base(const __bf16* P, int mt, int lane) {
  return P + (((size_t)mt * KT8) << 9) + lane * 16;
}
__device__ __forceinline__ v16bf ld_apk_at(const __bf16* base, int kt) {
  return *(const v16bf*)(base + kt * 512);
}

// D-fragment store into row-major bf16 [16][rs] buffer; base fixed per lane, v -> immediate.
__device__ __forceinline__ __bf16* d_st_base(__bf16* buf, int lane, int rs) {
  return buf + ((lane >> 4) * 8) * rs + (lane & 15);
}
__device__ __forceinline__ void st_d_at(__bf16* base, int nt, const v8f c, int rs) {
  #pragma unroll
  for (int v = 0; v < 8; ++v) base[v * rs + nt * 16] = (__bf16)c[v];
}

__device__ __forceinline__ float sigmoidf_(float x) { return 1.0f / (1.0f + __expf(-x)); }

// ---------------- Packing kernels ----------------

__global__ void pack_b_kernel(const float* __restrict__ src, __bf16* __restrict__ dst,
                              int Ntiles, int Ktiles, int rowStride, int colOff, int nLimit)
{
  int id = blockIdx.x * blockDim.x + threadIdx.x;
  int total = Ntiles * Ktiles * 32;
  if (id >= total) return;
  int lane = id & 31;
  int tile = id >> 5;
  int kt = tile % Ktiles;
  int nt = tile / Ktiles;
  int n  = nt * 16 + (lane & 15);
  int k0 = kt * 32 + (lane >> 4) * 8;
  __bf16* d = dst + (size_t)tile * 512;
  if (n < nLimit) {
    const float* s = src + (size_t)n * rowStride + colOff + k0;
    #pragma unroll
    for (int j = 0; j < 8; ++j) d[lane * 8 + j]       = (__bf16)s[j];
    #pragma unroll
    for (int j = 0; j < 8; ++j) d[256 + lane * 8 + j] = (__bf16)s[16 + j];
  } else {
    #pragma unroll
    for (int j = 0; j < 8; ++j) { d[lane * 8 + j] = (__bf16)0.f; d[256 + lane * 8 + j] = (__bf16)0.f; }
  }
}

__global__ void pack_a_kernel(const float* __restrict__ src, __bf16* __restrict__ dst,
                              int Mtiles, int Ktiles, int rowStride, int mLimit)
{
  int id = blockIdx.x * blockDim.x + threadIdx.x;
  int total = Mtiles * Ktiles * 32;
  if (id >= total) return;
  int lane = id & 31;
  int tile = id >> 5;
  int kt = tile % Ktiles;
  int mt = tile / Ktiles;
  int m  = mt * 16 + (lane & 15);
  int g8 = (lane >> 4) * 8;
  __bf16* d = dst + (size_t)tile * 512 + lane * 16;
  if (m < mLimit) {
    const float* s = src + (size_t)m * rowStride + kt * 32 + g8;
    #pragma unroll
    for (int j = 0; j < 8; ++j) d[j]     = (__bf16)s[j];
    #pragma unroll
    for (int j = 0; j < 8; ++j) d[8 + j] = (__bf16)s[16 + j];
  } else {
    #pragma unroll
    for (int j = 0; j < 16; ++j) d[j] = (__bf16)0.f;
  }
}

__global__ void add_bias_kernel(const float* a, const float* b, float* o, int n) {
  int i = blockIdx.x * blockDim.x + threadIdx.x;
  if (i < n) o[i] = a[i] + b[i];
}

__global__ void prep_scalars_kernel(const float* __restrict__ times, const int* __restrict__ types,
                                    float* __restrict__ dtPrev, float* __restrict__ scA,
                                    int* __restrict__ tpA)
{
  int i = blockIdx.x * blockDim.x + threadIdx.x;
  if (i >= BB * TT) return;
  int b = i >> 9, t = i & (TT - 1);
  const float* tb = times + (size_t)b * TT;
  float dt = tb[t] - (t ? tb[t - 1] : 0.f);
  scA[i] = fmaxf(dt, 0.001f);
  dtPrev[i] = (t >= 1) ? (tb[t - 1] - (t >= 2 ? tb[t - 2] : 0.f)) : 0.f;
  tpA[i] = (t >= 1) ? types[(size_t)b * TT + (t - 1)] : 500;   // row 500 of Ep = pure bias
}

// Generic packed-A x packed-B WMMA GEMM (+ per-column bias) -> f32 row-major C. K fixed = 256.
__global__ void __launch_bounds__(128)
gemm_pp_kernel(const __bf16* __restrict__ A, const __bf16* __restrict__ Bm,
               const float* __restrict__ bias, float* __restrict__ C,
               int Mtiles, int Ntiles, int ldc)
{
  int gw = (int)((blockIdx.x * blockDim.x + threadIdx.x) >> 5);
  int lane = threadIdx.x & 31;
  if (gw >= Mtiles * Ntiles) return;       // wave-uniform exit: EXEC stays all-ones for WMMA
  int mt = gw / Ntiles, nt = gw % Ntiles;
  const __bf16* aBase = a_pk_base(A, mt, lane);
  const __bf16* bBase = b_base(Bm, nt, lane);
  v8f c0 = {}, c1 = {};
  #pragma unroll
  for (int kt = 0; kt < KT8; kt += 2) {
    c0 = wmma_bf(ld_apk_at(aBase, kt),     ld_b_at(bBase, kt),     c0);
    c1 = wmma_bf(ld_apk_at(aBase, kt + 1), ld_b_at(bBase, kt + 1), c1);
  }
  int n  = nt * 16 + (lane & 15);
  float bv = bias ? bias[n] : 0.f;
  int mb = (lane >> 4) * 8;
  #pragma unroll
  for (int v = 0; v < 8; ++v)
    C[(size_t)(mt * 16 + mb + v) * ldc + n] = c0[v] + c1[v] + bv;
}

// ---------------- Sequential scan: GRU + RK4 ODE ----------------
// grid = 8 workgroups (one 16-row batch slice each), 512 threads = 16 waves.
// Wave w owns hidden columns [16w,16w+16) and gh n-tiles {w, 16+w, 32+w}.
// ODE weights (256KB packed bf16) live in the 320KB WGP LDS; w_hh streams from L2.
// All fragment base pointers are loop-invariant -> loads use immediate offsets only.
#define SCAN_W1_OFF  0
#define SCAN_W2_OFF  65536
#define SCAN_YB_OFF  131072
#define SCAN_UB_OFF  (131072 + 16 * ARS)
#define SCAN_SMEM_HALFS (131072 + 2 * 16 * ARS)

__global__ void __launch_bounds__(512)
scan_kernel(const __bf16* __restrict__ PWhh, const __bf16* __restrict__ PW1,
            const __bf16* __restrict__ PW2,  const float* __restrict__ Ep,
            const float* __restrict__ ob1,   const float* __restrict__ ob2,
            const float* __restrict__ wih,   const float* __restrict__ dtPrev,
            const float* __restrict__ scArr, const int* __restrict__ tpArr,
            __bf16* __restrict__ Hneg)
{
  extern __shared__ __bf16 smem[];
  __bf16* W1L = smem + SCAN_W1_OFF;   // 16x8 tiles (128KB)
  __bf16* W2L = smem + SCAN_W2_OFF;   // 16x8 tiles (128KB)
  __bf16* yB  = smem + SCAN_YB_OFF;   // [16][ARS] activations
  __bf16* uB  = smem + SCAN_UB_OFF;   // [16][ARS] tanh intermediate
  __shared__ float scL[16], dtL[16];
  __shared__ int   tpL[16];

  const int tid  = threadIdx.x;
  const int lane = tid & 31;
  const int w    = tid >> 5;
  const int g    = blockIdx.x;

  // cooperative fill: ODE weights -> LDS (once; each read 4x per step)
  for (int i = tid * 8; i < 16 * KT8 * 512; i += 512 * 8) {
    *(v8bf*)(W1L + i) = *(const v8bf*)(PW1 + i);
    *(v8bf*)(W2L + i) = *(const v8bf*)(PW2 + i);
  }
  for (int i = tid; i < 16 * ARS; i += 512) yB[i] = (__bf16)0.f;   // h0 = 0

  // ---- loop-invariant bases ----
  const __bf16* bR  = b_base(PWhh,      w, lane);
  const __bf16* bZ  = b_base(PWhh, 16 + w, lane);
  const __bf16* bN  = b_base(PWhh, 32 + w, lane);
  const __bf16* bW1 = b_base(W1L, w, lane);
  const __bf16* bW2 = b_base(W2L, w, lane);
  const __bf16* aY  = a_lds_base(yB, lane, ARS);
  const __bf16* aU  = a_lds_base(uB, lane, ARS);
  __bf16*       dY  = d_st_base(yB, lane, ARS);
  __bf16*       dU  = d_st_base(uB, lane, ARS);

  const int myN   = w * 16 + (lane & 15);
  const float b1v = ob1[myN];
  const float b2v = ob2[myN];
  const float w0r = wih[(size_t)(      myN) * 257];   // w_ih[:,0] per gate column
  const float w0z = wih[(size_t)(256 + myN) * 257];
  const float w0n = wih[(size_t)(512 + myN) * 257];
  const int mb = (lane >> 4) * 8;
  const float* epR = Ep + myN;        // + tp*768 at use
  const float* epZ = Ep + 256 + myN;
  const float* epN = Ep + 512 + myN;

  // running pointers for per-step data
  const float* scP = scArr + (size_t)(g * 16 + tid) * TT;   // valid for tid<16
  const float* dtP_ = dtPrev + (size_t)(g * 16 + tid) * TT;
  const int*   tpP = tpArr + (size_t)(g * 16 + tid) * TT;
  __bf16* HnegP = Hneg + (((size_t)g * TT * 8 + w) << 9) + lane * 16;   // + t*4096 halfs
  const __bf16* hOut = yB + (lane & 15) * ARS + w * 32 + (lane >> 4) * 8; // A-frag src, k-tile w

  v8f hp = {0.f,0.f,0.f,0.f,0.f,0.f,0.f,0.f};
  __syncthreads();

  #pragma unroll 1
  for (int t = 0; t < TT; ++t) {
    if (tid < 16) {
      scL[tid] = scP[t];
      dtL[tid] = dtP_[t];
      tpL[tid] = tpP[t];
    }
    __syncthreads();

    float scF[8], dtF[8]; int tpF[8];
    #pragma unroll
    for (int v = 0; v < 8; ++v) { scF[v] = scL[mb+v]; dtF[v] = dtL[mb+v]; tpF[v] = tpL[mb+v]; }

    // ---- gh = h @ w_hh^T : 3 gate tiles per wave (w_hh streamed from L2) ----
    v8f gr = {}; v8f gz = {}; v8f gn = {};
    #pragma unroll
    for (int kt = 0; kt < KT8; ++kt) {
      v16bf a = ld_a_at(aY, kt);
      gr = wmma_bf(a, ld_b_at(bR, kt), gr);
      gz = wmma_bf(a, ld_b_at(bZ, kt), gz);
      gn = wmma_bf(a, ld_b_at(bN, kt), gn);
    }

    // ---- GRU elementwise (gi gathered from precomputed Ep) ----
    v8f hpos;
    #pragma unroll
    for (int v = 0; v < 8; ++v) {
      size_t ro = (size_t)tpF[v] * 768;
      float ir = epR[ro] + dtF[v] * w0r;
      float iz = epZ[ro] + dtF[v] * w0z;
      float in = epN[ro] + dtF[v] * w0n;
      float r = sigmoidf_(ir + gr[v]);
      float z = sigmoidf_(iz + gz[v]);
      float n = tanhf(in + r * gn[v]);
      hpos[v] = (1.f - z) * n + z * hp[v];
    }
    __syncthreads();                 // all gh reads of yB complete
    st_d_at(dY, w, hpos, ARS);       // y <- h_pos

    // ---- RK4: 4 stages, each = two dependent 16x256x256 GEMMs (weights in LDS) ----
    v8f k1 = {}, k2 = {}, k3 = {}, hnew = {};
    #pragma unroll 1
    for (int s = 0; s < 4; ++s) {
      __syncthreads();               // y ready in yB; uB free
      v8f u0 = {}, u1 = {};
      #pragma unroll
      for (int kt = 0; kt < KT8; kt += 2) {
        u0 = wmma_bf(ld_a_at(aY, kt),     ld_b_at(bW1, kt),     u0);
        u1 = wmma_bf(ld_a_at(aY, kt + 1), ld_b_at(bW1, kt + 1), u1);
      }
      v8f uf;
      #pragma unroll
      for (int v = 0; v < 8; ++v) uf[v] = tanhf(u0[v] + u1[v] + b1v);
      st_d_at(dU, w, uf, ARS);
      __syncthreads();               // uB ready; yB reads done -> safe to overwrite
      v8f p0 = {}, p1 = {};
      #pragma unroll
      for (int kt = 0; kt < KT8; kt += 2) {
        p0 = wmma_bf(ld_a_at(aU, kt),     ld_b_at(bW2, kt),     p0);
        p1 = wmma_bf(ld_a_at(aU, kt + 1), ld_b_at(bW2, kt + 1), p1);
      }
      v8f kk;
      #pragma unroll
      for (int v = 0; v < 8; ++v) kk[v] = (p0[v] + p1[v] + b2v) * scF[v];

      v8f y;
      if (s == 0) {
        k1 = kk;
        #pragma unroll
        for (int v = 0; v < 8; ++v) y[v] = hpos[v] + kk[v] * (1.f / 3.f);
      } else if (s == 1) {
        k2 = kk;
        #pragma unroll
        for (int v = 0; v < 8; ++v) y[v] = hpos[v] + kk[v] - k1[v] * (1.f / 3.f);
      } else if (s == 2) {
        k3 = kk;
        #pragma unroll
        for (int v = 0; v < 8; ++v) y[v] = hpos[v] + k1[v] - k2[v] + kk[v];
      } else {
        #pragma unroll
        for (int v = 0; v < 8; ++v)
          y[v] = hpos[v] + (k1[v] + 3.f * (k2[v] + k3[v]) + kk[v]) * 0.125f;
        hnew = y;
      }
      st_d_at(dY, w, y, ARS);        // next y (or h_neg at s==3)
    }
    hp = hnew;
    __syncthreads();                 // yB holds h_neg (bf16)

    // write h_neg as packed A tiles for the tail GEMMs (waves 0..7 -> k-tiles 0..7)
    if (w < 8) {
      v8bf lo = *(const v8bf*)hOut;
      v8bf hi = *(const v8bf*)(hOut + 16);
      *(v8bf*)HnegP       = lo;
      *(v8bf*)(HnegP + 8) = hi;
    }
    HnegP += 8 * 512;                // next timestep's tile block
    __syncthreads();
  }
}

// ---------------- Fused tail: q = tanh(H@q1+b1)@q2+b2 ; out = q@psi^T ----------------
// grid = 4096 (one 16-row M-tile), 256 threads = 8 waves.
__global__ void __launch_bounds__(256)
post_kernel(const __bf16* __restrict__ Hneg, const __bf16* __restrict__ PQ1,
            const __bf16* __restrict__ PQ2,  const __bf16* __restrict__ PPSI,
            const float* __restrict__ qb1,   const float* __restrict__ qb2,
            float* __restrict__ out)
{
  __shared__ __bf16 uB[16 * ARS];
  __shared__ __bf16 qB[16 * ARS];
  const int mt   = blockIdx.x;
  const int lane = threadIdx.x & 31;
  const int w    = threadIdx.x >> 5;
  const __bf16* aH = Hneg + ((size_t)mt << 12) + lane * 16;   // packed A, 8 k-tiles
  const __bf16* aU = a_lds_base(uB, lane, ARS);
  const __bf16* aQ = a_lds_base(qB, lane, ARS);
  __bf16*       dU = d_st_base(uB, lane, ARS);
  __bf16*       dQ = d_st_base(qB, lane, ARS);

  // stage 1: u = tanh(H @ q_w1^T + q_b1)  (16 n-tiles over 8 waves)
  #pragma unroll
  for (int nn = 0; nn < 2; ++nn) {
    int nt = w + nn * 8;
    const __bf16* bQ = b_base(PQ1, nt, lane);
    v8f c0 = {}, c1 = {};
    #pragma unroll
    for (int kt = 0; kt < KT8; kt += 2) {
      c0 = wmma_bf(ld_apk_at(aH, kt),     ld_b_at(bQ, kt),     c0);
      c1 = wmma_bf(ld_apk_at(aH, kt + 1), ld_b_at(bQ, kt + 1), c1);
    }
    float bb = qb1[nt * 16 + (lane & 15)];
    v8f u;
    #pragma unroll
    for (int v = 0; v < 8; ++v) u[v] = tanhf(c0[v] + c1[v] + bb);
    st_d_at(dU, nt, u, ARS);
  }
  __syncthreads();

  // stage 2: q = u @ q_w2^T + q_b2
  #pragma unroll
  for (int nn = 0; nn < 2; ++nn) {
    int nt = w + nn * 8;
    const __bf16* bQ = b_base(PQ2, nt, lane);
    v8f c0 = {}, c1 = {};
    #pragma unroll
    for (int kt = 0; kt < KT8; kt += 2) {
      c0 = wmma_bf(ld_a_at(aU, kt),     ld_b_at(bQ, kt),     c0);
      c1 = wmma_bf(ld_a_at(aU, kt + 1), ld_b_at(bQ, kt + 1), c1);
    }
    float bb = qb2[nt * 16 + (lane & 15)];
    v8f q;
    #pragma unroll
    for (int v = 0; v < 8; ++v) q[v] = c0[v] + c1[v] + bb;
    st_d_at(dQ, nt, q, ARS);
  }
  __syncthreads();

  // stage 3: logits = q @ psi^T (N padded to 512, store-masked to 500)
  const int bslice = mt >> 9;
  const int t      = mt & (TT - 1);
  #pragma unroll
  for (int nn = 0; nn < 4; ++nn) {
    int nt = w + nn * 8;
    const __bf16* bP = b_base(PPSI, nt, lane);
    v8f c0 = {}, c1 = {};
    #pragma unroll
    for (int kt = 0; kt < KT8; kt += 2) {
      c0 = wmma_bf(ld_a_at(aQ, kt),     ld_b_at(bP, kt),     c0);
      c1 = wmma_bf(ld_a_at(aQ, kt + 1), ld_b_at(bP, kt + 1), c1);
    }
    int n = nt * 16 + (lane & 15);
    if (n < 500) {
      int mbv = (lane >> 4) * 8;
      float* op = out + ((size_t)(bslice * 16 + mbv) * TT + t) * 500 + n;
      #pragma unroll
      for (int v = 0; v < 8; ++v) op[(size_t)v * TT * 500] = c0[v] + c1[v];
    }
  }
}

// ---------------- Host launcher ----------------

extern "C" void kernel_launch(void* const* d_in, const int* in_sizes, int n_in,
                              void* d_out, int out_size, void* d_ws, size_t ws_size,
                              hipStream_t stream)
{
  const float* times  = (const float*)d_in[0];
  const int*   types  = (const int*)  d_in[1];
  const float* embed  = (const float*)d_in[2];
  const float* w_ih   = (const float*)d_in[3];
  const float* w_hh   = (const float*)d_in[4];
  const float* b_ih   = (const float*)d_in[5];
  const float* b_hh   = (const float*)d_in[6];
  const float* ode_w1 = (const float*)d_in[7];
  const float* ode_b1 = (const float*)d_in[8];
  const float* ode_w2 = (const float*)d_in[9];
  const float* ode_b2 = (const float*)d_in[10];
  const float* q_w1   = (const float*)d_in[11];
  const float* q_b1   = (const float*)d_in[12];
  const float* q_w2   = (const float*)d_in[13];
  const float* q_b2   = (const float*)d_in[14];
  const float* psi_w  = (const float*)d_in[15];
  const float* psi_b  = (const float*)d_in[16];
  float* out = (float*)d_out;

  char* ws = (char*)d_ws;
  size_t off = 0;
  auto alloc = [&](size_t bytes) -> char* {
    char* p = ws + off;
    off = (off + bytes + 1023) & ~(size_t)1023;
    return p;
  };

  __bf16* PWhh = (__bf16*)alloc((size_t)48 * 8 * 512 * 2);
  __bf16* PW1  = (__bf16*)alloc((size_t)16 * 8 * 512 * 2);
  __bf16* PW2  = (__bf16*)alloc((size_t)16 * 8 * 512 * 2);
  __bf16* PQ1  = (__bf16*)alloc((size_t)16 * 8 * 512 * 2);
  __bf16* PQ2  = (__bf16*)alloc((size_t)16 * 8 * 512 * 2);
  __bf16* PIH  = (__bf16*)alloc((size_t)48 * 8 * 512 * 2);
  __bf16* PPSW = (__bf16*)alloc((size_t)16 * 8 * 512 * 2);
  __bf16* PAE  = (__bf16*)alloc((size_t)32 * 8 * 512 * 2);
  __bf16* PPSI = (__bf16*)alloc((size_t)32 * 8 * 512 * 2);
  float*  Ep   = (float*) alloc((size_t)512 * 768 * 4);
  float*  psiF = (float*) alloc((size_t)512 * 256 * 4);
  float*  bias2= (float*) alloc((size_t)768 * 4);
  float*  dtP  = (float*) alloc((size_t)BB * TT * 4);
  float*  scA  = (float*) alloc((size_t)BB * TT * 4);
  int*    tpA  = (int*)   alloc((size_t)BB * TT * 4);
  __bf16* Hneg = (__bf16*)alloc((size_t)4096 * 8 * 512 * 2);
  (void)ws_size; (void)in_sizes; (void)n_in; (void)out_size;

  // --- pack weights (bf16, WMMA tile layout) ---
  pack_b_kernel<<<(48*8*32 + 255)/256, 256, 0, stream>>>(w_hh,   PWhh, 48, 8, 256, 0, 768);
  pack_b_kernel<<<(16*8*32 + 255)/256, 256, 0, stream>>>(ode_w1, PW1,  16, 8, 256, 0, 256);
  pack_b_kernel<<<(16*8*32 + 255)/256, 256, 0, stream>>>(ode_w2, PW2,  16, 8, 256, 0, 256);
  pack_b_kernel<<<(16*8*32 + 255)/256, 256, 0, stream>>>(q_w1,   PQ1,  16, 8, 256, 0, 256);
  pack_b_kernel<<<(16*8*32 + 255)/256, 256, 0, stream>>>(q_w2,   PQ2,  16, 8, 256, 0, 256);
  pack_b_kernel<<<(48*8*32 + 255)/256, 256, 0, stream>>>(w_ih,   PIH,  48, 8, 257, 1, 768);
  pack_b_kernel<<<(16*8*32 + 255)/256, 256, 0, stream>>>(psi_w,  PPSW, 16, 8, 256, 0, 256);
  pack_a_kernel<<<(32*8*32 + 255)/256, 256, 0, stream>>>(embed,  PAE,  32, 8, 256, 500);
  add_bias_kernel<<<3, 256, 0, stream>>>(b_ih, b_hh, bias2, 768);
  prep_scalars_kernel<<<(BB*TT + 255)/256, 256, 0, stream>>>(times, types, dtP, scA, tpA);

  // --- Ep = embed @ w_ih[:,1:]^T + (b_ih + b_hh), padded rows -> pure bias (t=0 case) ---
  gemm_pp_kernel<<<(32*48 + 3)/4, 128, 0, stream>>>(PAE, PIH, bias2, Ep, 32, 48, 768);
  // --- psi = embed @ psi_w^T + psi_b, then pack as B operand (padded to 512 rows) ---
  gemm_pp_kernel<<<(32*16 + 3)/4, 128, 0, stream>>>(PAE, PPSW, psi_b, psiF, 32, 16, 256);
  pack_b_kernel<<<(32*8*32 + 255)/256, 256, 0, stream>>>(psiF, PPSI, 32, 8, 256, 0, 512);

  // --- sequential scan (8 independent batch slices, ~272KB LDS: ODE weights resident) ---
  const size_t scanSmem = (size_t)SCAN_SMEM_HALFS * sizeof(__bf16);   // < 320KB WGP LDS
  hipFuncSetAttribute((const void*)scan_kernel, hipFuncAttributeMaxDynamicSharedMemorySize,
                      (int)scanSmem);
  scan_kernel<<<8, 512, scanSmem, stream>>>(PWhh, PW1, PW2, Ep, ode_b1, ode_b2, w_ih,
                                            dtP, scA, tpA, Hneg);

  // --- fused projection tail ---
  post_kernel<<<4096, 256, 0, stream>>>(Hneg, PQ1, PQ2, PPSI, q_b1, q_b2, out);
}
